// TransformerBlock_32727650796061
// MI455X (gfx1250) — matrix operand order repaired and checked by reading, not compiled, not run on hardware
//
#include <hip/hip_runtime.h>
#include <math.h>

// ---------------- constants ----------------
#define Bx 4
#define Cc 256
#define Nn 4096            // 16*16*16 tokens
#define NHh 8
#define HDd 32
#define Pp 64
#define BN_TOK (Bx*Nn)     // 16384

typedef __attribute__((ext_vector_type(16))) __bf16 v16bf;
typedef __attribute__((ext_vector_type(8)))  float  v8f;

// ---------------- bf16 helpers ----------------
__device__ __forceinline__ __bf16 f2bf(float f) {
    union { float f; unsigned u; } x; x.f = f;
    unsigned r = x.u + 0x7FFFu + ((x.u >> 16) & 1u);
    union { unsigned short s; __bf16 b; } y; y.s = (unsigned short)(r >> 16);
    return y.b;
}
__device__ __forceinline__ float bf2f(__bf16 b) {
    union { unsigned short s; __bf16 b; } y; y.b = b;
    union { unsigned u; float f; } x; x.u = ((unsigned)y.s) << 16;
    return x.f;
}

// ---------------- WMMA wrappers ----------------
__device__ __forceinline__ v8f wmma_bf(v16bf a, v16bf b, v8f c) {
    return __builtin_amdgcn_wmma_f32_16x16x32_bf16(false, a, false, b, (short)0, c, false, false);
}

// A fragment: logical A[m0..m0+15, k0..k0+31], row-major with row stride rs (cs=1).
// Per ISA 16-bit A 16x32 layout; per lane this is two contiguous 8x bf16 chunks.
__device__ __forceinline__ v16bf load_a(const __bf16* A, int rs, int m0, int k0) {
    const int lane = threadIdx.x & 31;
    const __bf16* rowp = A + (size_t)(m0 + (lane & 15)) * rs + k0 + ((lane & 16) ? 8 : 0);
    v16bf r;
#pragma unroll
    for (int e = 0; e < 16; ++e)
        r[e] = rowp[((e & 8) ? 16 : 0) + (e & 7)];
    return r;
}

// A fragment with per-lane row pointer + validity (conv zero-padding).
__device__ __forceinline__ v16bf load_a_row(const __bf16* rowp0, int k0, bool ok) {
    const int lane = threadIdx.x & 31;
    const __bf16* rowp = rowp0 + k0 + ((lane & 16) ? 8 : 0);
    v16bf r;
#pragma unroll
    for (int e = 0; e < 16; ++e)
        r[e] = ok ? rowp[((e & 8) ? 16 : 0) + (e & 7)] : f2bf(0.0f);
    return r;
}

// B fragment: logical B[k0..k0+31, n0..n0+15], K-CONTIGUOUS storage:
// element (k,n) at Bp[n*cs + k].  Per lane: two contiguous 8x bf16 chunks (same k order as load_a).
__device__ __forceinline__ v16bf load_b(const __bf16* Bp, int cs, int k0, int n0) {
    const int lane = threadIdx.x & 31;
    const __bf16* p = Bp + (size_t)(n0 + (lane & 15)) * cs + k0 + ((lane & 16) ? 8 : 0);
    v16bf r;
#pragma unroll
    for (int e = 0; e < 16; ++e) r[e] = p[((e & 8) ? 16 : 0) + (e & 7)];
    return r;
}

// ---------------- kernel 1: LayerNorm + token-major residual ----------------
__global__ void k_ln(const float* __restrict__ x, const float* __restrict__ lnw,
                     const float* __restrict__ lnb, float* __restrict__ attn,
                     __bf16* __restrict__ xn) {
    __shared__ float red[256];
    const int t = threadIdx.x;
    const int tok = blockIdx.x;
    const int b = tok >> 12, n = tok & 4095;
    float v = x[(size_t)b * (Cc * Nn) + (size_t)t * Nn + n];
    attn[(size_t)tok * Cc + t] = v;
    red[t] = v; __syncthreads();
    for (int s = 128; s > 0; s >>= 1) { if (t < s) red[t] += red[t + s]; __syncthreads(); }
    float mu = red[0] * (1.0f / Cc); __syncthreads();
    float d = v - mu;
    red[t] = d * d; __syncthreads();
    for (int s = 128; s > 0; s >>= 1) { if (t < s) red[t] += red[t + s]; __syncthreads(); }
    float var = red[0] * (1.0f / Cc);
    float xh = d * rsqrtf(var + 1e-5f) * lnw[t] + lnb[t];
    xn[(size_t)tok * Cc + t] = f2bf(xh);
}

// ---------------- packing kernels ----------------
__global__ void k_cvt(const float* __restrict__ s, __bf16* __restrict__ d, int n) {
    int i = blockIdx.x * 256 + threadIdx.x;
    if (i < n) d[i] = f2bf(s[i]);
}
// EF [n][p] -> ef_t [p][n]  (token index contiguous)
__global__ void k_pack_efT(const float* __restrict__ ef, __bf16* __restrict__ eft) {
    int i = blockIdx.x * 256 + threadIdx.x;          // 262144
    int n = i >> 6, p = i & 63;
    eft[(size_t)p * Nn + n] = f2bf(ef[i]);
}
// conv w [co][ci][27] -> wT [co][kk*256+ci]   (K-contiguous per co)
__global__ void k_pack_convT(const float* __restrict__ w, __bf16* __restrict__ wt) {
    int i = blockIdx.x * 256 + threadIdx.x;          // 256*256*27
    int co = i / (256 * 27);
    int r  = i - co * (256 * 27);
    int ci = r / 27, kk = r - ci * 27;
    wt[(size_t)co * 6912 + kk * 256 + ci] = f2bf(w[i]);
}

// ---------------- qkvv GEMM (32x32 per wave), scatter to attention layouts ------
__global__ void k_qkvv(const __bf16* __restrict__ xn, const __bf16* __restrict__ wtb,
                       __bf16* __restrict__ sbuf) {
    int wave = (blockIdx.x * blockDim.x + threadIdx.x) >> 5;   // 16384 waves
    int ntg = wave & 31, mt = wave >> 5;                        // 512 x 32
    int m0 = mt << 5, n0 = ntg << 5;
    v8f acc[2][2] = {};
#pragma unroll
    for (int k0 = 0; k0 < Cc; k0 += 32) {
        v16bf a0 = load_a(xn, Cc, m0, k0);
        v16bf a1 = load_a(xn, Cc, m0 + 16, k0);
        v16bf b0 = load_b(wtb, Cc, k0, n0);        // B[k=c,col=j]=qkvv_w[j][c]
        v16bf b1 = load_b(wtb, Cc, k0, n0 + 16);
        acc[0][0] = wmma_bf(a0, b0, acc[0][0]);
        acc[0][1] = wmma_bf(a0, b1, acc[0][1]);
        acc[1][0] = wmma_bf(a1, b0, acc[1][0]);
        acc[1][1] = wmma_bf(a1, b1, acc[1][1]);
    }
    const int lane = threadIdx.x & 31;
    const int mb = (lane & 16) ? 8 : 0;
#pragma unroll
    for (int jt = 0; jt < 2; ++jt) {
        int j = n0 + jt * 16 + (lane & 15);
        int s = j >> 8, hh = (j >> 5) & 7, dd = j & 31;
#pragma unroll
        for (int it = 0; it < 2; ++it)
#pragma unroll
            for (int r = 0; r < 8; ++r) {
                int m = m0 + it * 16 + mb + r;
                int b = m >> 12, n = m & 4095;
                size_t idx;
                if (s == 2)   // v_ca: token-major [b][h][n][d] inside slot 2
                    idx = (size_t)2 * Bx * NHh * HDd * Nn
                        + ((size_t)(b * NHh + hh) * Nn + n) * HDd + dd;
                else          // q/k/v_sa: [s][b][h][d][n]
                    idx = ((((size_t)s * Bx + b) * NHh + hh) * HDd + dd) * Nn + n;
                sbuf[idx] = f2bf(acc[it][jt][r]);
            }
    }
}

// ---------------- K/V_SA projections with EF (K=4096) ----------------
// kp stored TRANSPOSED [bh][p][d]; vp stored [bh][d][p].
__global__ void k_proj(const __bf16* __restrict__ sbuf, const __bf16* __restrict__ eft,
                       __bf16* __restrict__ kpt, __bf16* __restrict__ vp) {
    int wave = (blockIdx.x * blockDim.x + threadIdx.x) >> 5;   // 512 waves
    int which = wave >> 8;                                     // 0 -> k (s=1), 1 -> v_sa (s=3)
    int rem = wave & 255;
    int bh = rem >> 3, tile = rem & 7;
    int mt = tile >> 2, nt = tile & 3;
    int s = which ? 3 : 1;
    const __bf16* A = sbuf + (size_t)(s * 32 + bh) * HDd * Nn;   // [32 x 4096]
    int m0 = mt * 16, n0 = nt * 16;
    v8f acc = {};
    for (int k0 = 0; k0 < Nn; k0 += 32) {
        v16bf a = load_a(A, Nn, m0, k0);
        v16bf bb = load_b(eft, Nn, k0, n0);        // B[k=tok,col=p]=eft[p][tok]
        acc = wmma_bf(a, bb, acc);
    }
    const int lane = threadIdx.x & 31;
    const int p = n0 + (lane & 15);
    const int mb = (lane & 16) ? 8 : 0;
#pragma unroll
    for (int r = 0; r < 8; ++r) {
        int d = m0 + mb + r;
        if (which) vp [(size_t)bh * (HDd * Pp) + (size_t)d * Pp + p] = f2bf(acc[r]);
        else       kpt[(size_t)bh * (HDd * Pp) + (size_t)p * HDd + d] = f2bf(acc[r]);
    }
}

// ---------------- cosine-normalize q,k rows; emit token-major qn_t ----------------
__global__ void k_norm(__bf16* __restrict__ sbuf, __bf16* __restrict__ qnt) {
    __shared__ float red[256];
    const int row = blockIdx.x;                    // 0..2047
    const int s = row >> 10, bh = (row >> 5) & 31, d = row & 31;
    const size_t base = (size_t)row * Nn;
    const int t = threadIdx.x;
    float ss = 0.f;
    for (int i = t; i < Nn; i += 256) { float v = bf2f(sbuf[base + i]); ss += v * v; }
    red[t] = ss; __syncthreads();
    for (int st = 128; st > 0; st >>= 1) { if (t < st) red[t] += red[t + st]; __syncthreads(); }
    float inv = 1.0f / fmaxf(sqrtf(red[0]), 1e-12f);
    for (int i = t; i < Nn; i += 256) {
        float v = bf2f(sbuf[base + i]) * inv;
        __bf16 h = f2bf(v);
        sbuf[base + i] = h;
        if (s == 0) qnt[((size_t)bh * Nn + i) * HDd + d] = h;
    }
}

// ---------------- channel-attention logits + softmax (32x32, K=4096) -----
__global__ void k_attn_ca(const __bf16* __restrict__ sbuf, const float* __restrict__ temp1,
                          __bf16* __restrict__ aca) {
    __shared__ float tile[32 * 32];
    const int bh = blockIdx.x, h = bh & 7;
    const __bf16* qn = sbuf + (size_t)(0 * 32 + bh) * HDd * Nn;
    const __bf16* kn = sbuf + (size_t)(1 * 32 + bh) * HDd * Nn;
    v8f acc[2][2] = {};
    for (int k0 = 0; k0 < Nn; k0 += 32) {
        v16bf a0 = load_a(qn, Nn, 0, k0);
        v16bf a1 = load_a(qn, Nn, 16, k0);
        v16bf b0 = load_b(kn, Nn, k0, 0);          // B[k=tok,col=e]=kn[e][tok]
        v16bf b1 = load_b(kn, Nn, k0, 16);
        acc[0][0] = wmma_bf(a0, b0, acc[0][0]);
        acc[0][1] = wmma_bf(a0, b1, acc[0][1]);
        acc[1][0] = wmma_bf(a1, b0, acc[1][0]);
        acc[1][1] = wmma_bf(a1, b1, acc[1][1]);
    }
    const float t1 = temp1[h];
    const int lane = threadIdx.x;
    const int nn = lane & 15, mb = (lane & 16) ? 8 : 0;
#pragma unroll
    for (int mt = 0; mt < 2; ++mt)
#pragma unroll
        for (int nt = 0; nt < 2; ++nt)
#pragma unroll
            for (int r = 0; r < 8; ++r)
                tile[(mt * 16 + mb + r) * 32 + nt * 16 + nn] = acc[mt][nt][r] * t1;
    __syncthreads();
    const int m = lane;
    float mx = -1e30f;
    for (int e = 0; e < 32; ++e) mx = fmaxf(mx, tile[m * 32 + e]);
    float sum = 0.f;
    for (int e = 0; e < 32; ++e) { float ex = __expf(tile[m * 32 + e] - mx); sum += ex; tile[m * 32 + e] = ex; }
    float inv = 1.0f / sum;
    for (int e = 0; e < 32; ++e)
        aca[(size_t)bh * 1024 + m * 32 + e] = f2bf(tile[m * 32 + e] * inv);
}

// ---------------- x_CA = attn_CA @ V_CA, gamma-scaled add into attn -------
__global__ void k_xca(const __bf16* __restrict__ aca, const __bf16* __restrict__ sbuf,
                      const float* __restrict__ gamma, float* __restrict__ attn) {
    int wave = (blockIdx.x * blockDim.x + threadIdx.x) >> 5;   // 16384 waves
    int nt = wave & 255, mt = (wave >> 8) & 1, bh = wave >> 9;
    int b = bh >> 3, h = bh & 7;
    const __bf16* A = aca + (size_t)bh * 1024;                 // [32 x 32]
    const __bf16* V = sbuf + (size_t)2 * Bx * NHh * HDd * Nn
                    + (size_t)bh * Nn * HDd;                   // v_ca token-major [n][d]
    int m0 = mt * 16, n0 = nt * 16;
    v8f acc = {};
    acc = wmma_bf(load_a(A, 32, m0, 0), load_b(V, HDd, 0, n0), acc);  // B[k=e,col=n]=V[n][e]
    const int lane = threadIdx.x & 31;
    const int n = n0 + (lane & 15);
    const int mb = (lane & 16) ? 8 : 0;
#pragma unroll
    for (int r = 0; r < 8; ++r) {
        int d = m0 + mb + r;
        int c = h * HDd + d;
        size_t idx = ((size_t)b * Nn + n) * Cc + c;
        attn[idx] += gamma[c] * acc[r];
    }
}

// ---------------- fused spatial attention (logits -> softmax -> @V) -------
__global__ void k_xsa(const __bf16* __restrict__ qnt, const __bf16* __restrict__ kpt,
                      const __bf16* __restrict__ vp, const float* __restrict__ temp2,
                      const float* __restrict__ gamma, float* __restrict__ attn) {
    __shared__ float  lg[8][16 * 64];
    __shared__ __bf16 lb[8][16 * 64];
    const int wid = threadIdx.x >> 5, lane = threadIdx.x & 31;
    const int wave = blockIdx.x * 8 + wid;          // 8192 waves
    const int nt = wave & 255, bh = wave >> 8;
    const int b = bh >> 3, h = bh & 7;
    const __bf16* QT = qnt + (size_t)bh * Nn * HDd;        // [token][d]
    const __bf16* KP = kpt + (size_t)bh * (HDd * Pp);      // [p][d]
    const __bf16* VP = vp  + (size_t)bh * (HDd * Pp);      // [d][p]
    const int n0 = nt * 16;
    const float t2 = temp2[h];

    // logits: A = qn_t (16 tokens x 32 d), B[k=d,col=p]=KP[p][d]; K=32, 4 col tiles
    v16bf a = load_a(QT, HDd, n0, 0);
    v8f acc[4];
#pragma unroll
    for (int ct = 0; ct < 4; ++ct) {
        v8f z = {};
        acc[ct] = wmma_bf(a, load_b(KP, HDd, 0, ct * 16), z);
    }
    {
        const int nn = lane & 15, mb = (lane & 16) ? 8 : 0;
#pragma unroll
        for (int ct = 0; ct < 4; ++ct)
#pragma unroll
            for (int r = 0; r < 8; ++r)
                lg[wid][(mb + r) * 64 + ct * 16 + nn] = acc[ct][r] * t2;
    }
    __syncthreads();
    if (lane < 16) {
        float mx = -1e30f;
        for (int p = 0; p < Pp; ++p) mx = fmaxf(mx, lg[wid][lane * 64 + p]);
        float sum = 0.f;
        for (int p = 0; p < Pp; ++p) { float e = __expf(lg[wid][lane * 64 + p] - mx); sum += e; lg[wid][lane * 64 + p] = e; }
        float inv = 1.0f / sum;
        for (int p = 0; p < Pp; ++p) lb[wid][lane * 64 + p] = f2bf(lg[wid][lane * 64 + p] * inv);
    }
    __syncthreads();
    // x_SA tile: A = probs [16 x 64], B[k=p,col=d]=VP[d][p]; K=64
    v8f xacc[2] = {};
    for (int k0 = 0; k0 < Pp; k0 += 32) {
        v16bf ap = load_a(&lb[wid][0], 64, 0, k0);
        xacc[0] = wmma_bf(ap, load_b(VP, Pp, k0, 0), xacc[0]);
        xacc[1] = wmma_bf(ap, load_b(VP, Pp, k0, 16), xacc[1]);
    }
    {
        const int col = lane & 15, mb = (lane & 16) ? 8 : 0;
#pragma unroll
        for (int ti = 0; ti < 2; ++ti)
#pragma unroll
            for (int r = 0; r < 8; ++r) {
                int n = n0 + mb + r;
                int dd = ti * 16 + col;
                int f = (dd * NHh + h) * Nn + n;       // torch-faithful quirky reshape
                int tok = f >> 8, c = f & 255;
                size_t idx = ((size_t)b * Nn + tok) * Cc + c;
                attn[idx] += gamma[c] * xacc[ti][r];
            }
    }
}

// ---------------- 3x3x3 conv as implicit WMMA GEMM + BN + LReLU (+res) ----
__global__ void k_conv3(const __bf16* __restrict__ in, const __bf16* __restrict__ wt,
                        const float* __restrict__ bw, const float* __restrict__ bb,
                        const float* __restrict__ bm, const float* __restrict__ bv,
                        const float* __restrict__ res, __bf16* __restrict__ out, int mode) {
    int wave = (blockIdx.x * blockDim.x + threadIdx.x) >> 5;   // 4096 waves
    int cg = wave & 3, mt = wave >> 2;
    int m0 = mt << 4;
    int b = m0 >> 12, n = m0 & 4095;
    int z = n >> 8, y = (n >> 4) & 15;
    const int lane = threadIdx.x & 31;
    v8f acc[4] = {};
    for (int kk = 0; kk < 27; ++kk) {
        int dz = kk / 9 - 1, dy = (kk / 3) % 3 - 1, dx = kk % 3 - 1;
        int zz = z + dz, yy = y + dy;
        int xx = (lane & 15) + dx;
        bool ok = (zz >= 0) && (zz < 16) && (yy >= 0) && (yy < 16) && (xx >= 0) && (xx < 16);
        int zc = min(max(zz, 0), 15), yc = min(max(yy, 0), 15), xc = min(max(xx, 0), 15);
        const __bf16* rowp = in + (((size_t)b * Nn + (zc << 8) + (yc << 4) + xc) << 8);
#pragma unroll
        for (int k0 = 0; k0 < Cc; k0 += 32) {
            v16bf af = load_a_row(rowp, k0, ok);
#pragma unroll
            for (int ct = 0; ct < 4; ++ct)
                acc[ct] = wmma_bf(af, load_b(wt, 6912, kk * 256 + k0, cg * 64 + ct * 16), acc[ct]);
        }
    }
    const int mb = (lane & 16) ? 8 : 0;
#pragma unroll
    for (int ct = 0; ct < 4; ++ct) {
        int co = cg * 64 + ct * 16 + (lane & 15);
        float scale = bw[co] * rsqrtf(bv[co] + 1e-5f);
        float shift = bb[co] - bm[co] * scale;
#pragma unroll
        for (int r = 0; r < 8; ++r) {
            int m = m0 + mb + r;
            float val = acc[ct][r] * scale + shift;
            if (mode == 2) val += res[(size_t)m * Cc + co];
            val = (val >= 0.f) ? val : 0.01f * val;
            out[(size_t)m * Cc + co] = f2bf(val);
        }
    }
}

// ---------------- 1x1x1 conv8 (32x32 per wave) + residuals -> d_out [B,C,N] ------
__global__ void k_conv8(const __bf16* __restrict__ in, const __bf16* __restrict__ w8b,
                        const float* __restrict__ cb, const float* __restrict__ attn,
                        float* __restrict__ yout) {
    int wave = (blockIdx.x * blockDim.x + threadIdx.x) >> 5;   // 4096 waves
    int ntg = wave & 7, mt = wave >> 3;                         // 512 x 8
    int m0 = mt << 5, n0 = ntg << 5;
    v8f acc[2][2] = {};
#pragma unroll
    for (int k0 = 0; k0 < Cc; k0 += 32) {
        v16bf a0 = load_a(in, Cc, m0, k0);
        v16bf a1 = load_a(in, Cc, m0 + 16, k0);
        v16bf b0 = load_b(w8b, Cc, k0, n0);        // B[k=ci,col=co]=w8[co][ci]
        v16bf b1 = load_b(w8b, Cc, k0, n0 + 16);
        acc[0][0] = wmma_bf(a0, b0, acc[0][0]);
        acc[0][1] = wmma_bf(a0, b1, acc[0][1]);
        acc[1][0] = wmma_bf(a1, b0, acc[1][0]);
        acc[1][1] = wmma_bf(a1, b1, acc[1][1]);
    }
    const int lane = threadIdx.x & 31;
    const int mb = (lane & 16) ? 8 : 0;
#pragma unroll
    for (int jt = 0; jt < 2; ++jt) {
        int co = n0 + jt * 16 + (lane & 15);
        float bias = cb[co];
#pragma unroll
        for (int it = 0; it < 2; ++it)
#pragma unroll
            for (int r = 0; r < 8; ++r) {
                int m = m0 + it * 16 + mb + r;
                int b = m >> 12, n = m & 4095;
                yout[(size_t)b * (Cc * Nn) + (size_t)co * Nn + n] =
                    attn[(size_t)m * Cc + co] + acc[it][jt][r] + bias;
            }
    }
}

// ---------------- host launcher ----------------
extern "C" void kernel_launch(void* const* d_in, const int* in_sizes, int n_in,
                              void* d_out, int out_size, void* d_ws, size_t ws_size,
                              hipStream_t stream) {
    (void)in_sizes; (void)n_in; (void)out_size; (void)ws_size;
    const float* x       = (const float*)d_in[0];
    const float* ln_w    = (const float*)d_in[1];
    const float* ln_b    = (const float*)d_in[2];
    const float* gamma   = (const float*)d_in[3];
    const float* qkvv_w  = (const float*)d_in[4];
    const float* EF      = (const float*)d_in[5];
    const float* temp1   = (const float*)d_in[6];
    const float* temp2   = (const float*)d_in[7];
    const float* conv1_w = (const float*)d_in[8];
    const float* conv2_w = (const float*)d_in[9];
    const float* bn1_w   = (const float*)d_in[10];
    const float* bn1_b   = (const float*)d_in[11];
    const float* bn1_m   = (const float*)d_in[12];
    const float* bn1_v   = (const float*)d_in[13];
    const float* bn2_w   = (const float*)d_in[14];
    const float* bn2_b   = (const float*)d_in[15];
    const float* bn2_m   = (const float*)d_in[16];
    const float* bn2_v   = (const float*)d_in[17];
    const float* conv8_w = (const float*)d_in[18];
    const float* conv8_b = (const float*)d_in[19];
    float* yout = (float*)d_out;

    char* w = (char*)d_ws;
    float*  attn = (float*)w;   w += (size_t)BN_TOK * Cc * 4;             // 16 MB
    __bf16* xn   = (__bf16*)w;  w += (size_t)BN_TOK * Cc * 2;             // 8 MB (reused as out1)
    __bf16* wtb  = (__bf16*)w;  w += (size_t)Cc * 1024 * 2;               // 512 KB [j][c]
    __bf16* eft  = (__bf16*)w;  w += (size_t)Nn * Pp * 2;                 // 512 KB [p][n]
    __bf16* sbuf = (__bf16*)w;  w += (size_t)4 * Bx * NHh * HDd * Nn * 2; // 32 MB
    __bf16* qnt  = (__bf16*)w;  w += (size_t)Bx * NHh * Nn * HDd * 2;     // 8 MB [bh][n][d]
    __bf16* kpt  = (__bf16*)w;  w += (size_t)Bx * NHh * HDd * Pp * 2;     // 128 KB [bh][p][d]
    __bf16* vp   = (__bf16*)w;  w += (size_t)Bx * NHh * HDd * Pp * 2;     // 128 KB [bh][d][p]
    __bf16* aca  = (__bf16*)w;  w += (size_t)Bx * NHh * 32 * 32 * 2;      // 64 KB
    __bf16* w1t  = (__bf16*)w;  w += (size_t)27 * 256 * 256 * 2;          // 3.375 MB [co][k]
    __bf16* w2t  = (__bf16*)w;  w += (size_t)27 * 256 * 256 * 2;          // 3.375 MB
    __bf16* w8b  = (__bf16*)w;  w += (size_t)256 * 256 * 2;               // 128 KB [co][ci]
    __bf16* abf  = (__bf16*)w;  w += (size_t)BN_TOK * Cc * 2;             // 8 MB (reused as out2)
    __bf16* out1 = xn;    // xn dead after qkvv GEMM
    __bf16* out2 = abf;   // attn_bf dead after conv1

    k_ln<<<BN_TOK, 256, 0, stream>>>(x, ln_w, ln_b, attn, xn);
    k_cvt<<<1024, 256, 0, stream>>>(qkvv_w, wtb, Cc * 1024);
    k_pack_efT<<<1024, 256, 0, stream>>>(EF, eft);
    k_qkvv<<<2048, 256, 0, stream>>>(xn, wtb, sbuf);
    k_proj<<<64, 256, 0, stream>>>(sbuf, eft, kpt, vp);
    k_norm<<<2048, 256, 0, stream>>>(sbuf, qnt);
    k_attn_ca<<<32, 32, 0, stream>>>(sbuf, temp1, aca);
    k_xca<<<2048, 256, 0, stream>>>(aca, sbuf, gamma, attn);
    k_xsa<<<1024, 256, 0, stream>>>(qnt, kpt, vp, temp2, gamma, attn);
    k_cvt<<<BN_TOK, 256, 0, stream>>>(attn, abf, BN_TOK * Cc);
    k_pack_convT<<<6912, 256, 0, stream>>>(conv1_w, w1t);
    k_pack_convT<<<6912, 256, 0, stream>>>(conv2_w, w2t);
    k_cvt<<<256, 256, 0, stream>>>(conv8_w, w8b, 256 * 256);
    k_conv3<<<512, 256, 0, stream>>>(abf, w1t, bn1_w, bn1_b, bn1_m, bn1_v, attn, out1, 1);
    k_conv3<<<512, 256, 0, stream>>>(out1, w2t, bn2_w, bn2_b, bn2_m, bn2_v, attn, out2, 2);
    k_conv8<<<512, 256, 0, stream>>>(out2, w8b, conv8_b, attn, yout);
}
